// ENNModelWithSparsityControl_47502338294480
// MI455X (gfx1250) — matrix operand order repaired and checked by reading, not compiled, not run on hardware
//
#include <hip/hip_runtime.h>
#include <hip/hip_bf16.h>

typedef __attribute__((ext_vector_type(16))) __bf16 bf16x16;
typedef __attribute__((ext_vector_type(8)))  __bf16 bf16x8;
typedef __attribute__((ext_vector_type(8)))  float  f32x8;

union V16 { bf16x16 v; bf16x8 h[2]; };

#define WMMA_BF16(A, B, C) \
    __builtin_amdgcn_wmma_f32_16x16x32_bf16(false, (A), false, (B), (short)0, (C), false, false)

// ---------------------------------------------------------------------------
// K0: convert proj_w (4096x512) and x (16384x512) f32 -> bf16
// ---------------------------------------------------------------------------
__global__ void k_prep(const float* __restrict__ pw, __bf16* __restrict__ pbf,
                       const float* __restrict__ x, __bf16* __restrict__ xbf) {
    int i = blockIdx.x * 256 + threadIdx.x;
    if (i < 4096 * 512) pbf[i] = (__bf16)pw[i];
    if (i < 16384 * 512) xbf[i] = (__bf16)x[i];
}

// ---------------------------------------------------------------------------
// K2: projection GEMM  x3[b, c] = sum_k x[b,k] * proj_w[c,k] + proj_b[c]
//     Tile: 128 rows x 128 cols per WG; wave owns 16 cols x 128 rows.
//     x tile staged global->LDS with async B128 copies (ASYNCcnt path).
// ---------------------------------------------------------------------------
__global__ void k_proj(const __bf16* __restrict__ xbf, const __bf16* __restrict__ pwbf,
                       const float* __restrict__ pb, __bf16* __restrict__ x3bf) {
    extern __shared__ char smem[];
    __bf16* xs = (__bf16*)smem;           // [128][520] bf16 (padded rows), offset 0
    const int PAD = 520;

    const int tid  = threadIdx.x;
    const int wave = tid >> 5;
    const int lane = tid & 31;
    const int hi   = lane >> 4;
    const int lo   = lane & 15;
    const int m0   = blockIdx.x * 128;
    const int n0   = blockIdx.y * 128;

    // async-stage x tile (128 rows x 512 bf16) in 16B chunks, padded LDS rows
    // 8192 chunks total, 32 per thread
    for (int q = tid; q < 8192; q += 256) {
        int r  = q >> 6;            // row 0..127
        int c8 = (q & 63) << 3;     // col in elements, multiple of 8
        unsigned ldsoff = (unsigned)((r * PAD + c8) * 2);
        const __bf16* gp = xbf + (size_t)(m0 + r) * 512 + c8;
        asm volatile("global_load_async_to_lds_b128 %0, %1, off"
                     :: "v"(ldsoff), "v"(gp) : "memory");
    }
    asm volatile("s_wait_asynccnt 0x0" ::: "memory");
    __syncthreads();

    const int col = n0 + wave * 16 + lo;  // global output column this lane owns

    f32x8 acc[8];
    for (int mt = 0; mt < 8; ++mt)
        for (int r = 0; r < 8; ++r) acc[mt][r] = 0.0f;

    for (int kb = 0; kb < 512; kb += 32) {
        // B fragment from global bf16 proj_w: B[k, col] = proj_w[col, k]
        V16 bf;
        const __bf16* bp = pwbf + (size_t)col * 512 + kb + 8 * hi;
        bf.h[0] = *(const bf16x8*)(bp);
        bf.h[1] = *(const bf16x8*)(bp + 16);
        // all 8 A fragments first (one ds clause, one wait), then 8 WMMAs
        V16 af[8];
        for (int mt = 0; mt < 8; ++mt) {
            const __bf16* ap = xs + (mt * 16 + lo) * PAD + kb + 8 * hi;
            af[mt].h[0] = *(const bf16x8*)(ap);
            af[mt].h[1] = *(const bf16x8*)(ap + 16);
        }
        for (int mt = 0; mt < 8; ++mt)
            acc[mt] = WMMA_BF16(af[mt].v, bf.v, acc[mt]);
    }

    const float bias = pb[col];
    for (int mt = 0; mt < 8; ++mt) {
        for (int r = 0; r < 8; ++r) {
            int m = m0 + mt * 16 + r + 8 * hi;
            x3bf[(size_t)m * 4096 + col] = (__bf16)(acc[mt][r] + bias);
        }
    }
}

// ---------------------------------------------------------------------------
// K3: fused  mask -> mixing (x3) -> per-layer partial batch sums -> logits
//     One workgroup per 16 batch rows; x3 tile in LDS [s][b][n], n padded 136.
// ---------------------------------------------------------------------------
__global__ void k_fused(const __bf16* __restrict__ x3bf, const float* __restrict__ ent,
                        const float* __restrict__ mixing, const float* __restrict__ readout,
                        float* __restrict__ logits, float* __restrict__ part) {
    extern __shared__ char smem[];
    const int PAD = 136;
    __bf16* buf0 = (__bf16*)smem;          // 32*16*136 = 69632 elems
    __bf16* buf1 = buf0 + 69632;           // ping-pong
    __bf16* mixs = buf1 + 69632;           // [m][n] (transposed mixing), 128*136

    const int tid  = threadIdx.x;
    const int wave = tid >> 5;
    const int lane = tid & 31;
    const int hi   = lane >> 4;
    const int lo   = lane & 15;
    const int b0   = blockIdx.x * 16;

    // stage x3 tile: [16 b][4096 (n,s)] global -> [s][b][n] LDS (transpose)
    for (int i = tid; i < 16 * 4096; i += 256) {
        int b = i >> 12, c = i & 4095;
        int n = c >> 5, s = c & 31;
        buf0[(s * 16 + b) * PAD + n] = x3bf[(size_t)(b0 + b) * 4096 + c];
    }
    // stage mixing transposed: mixs[m][n] = mixing[n][m]
    for (int i = tid; i < 128 * 128; i += 256) {
        int n = i >> 7, m = i & 127;
        mixs[m * PAD + n] = (__bf16)mixing[i];
    }
    __syncthreads();

    __bf16* cur = buf0;
    __bf16* nxt = buf1;

    for (int layer = 0; layer < 3; ++layer) {
        // (a) elementwise sigmoid mask
        for (int p = tid; p < 4096; p += 256) {
            int n = p >> 5, s = p & 31;
            float mk = 1.0f / (1.0f + __expf(-ent[p]));
            for (int b = 0; b < 16; ++b) {
                int idx = (s * 16 + b) * PAD + n;
                cur[idx] = (__bf16)((float)cur[idx] * mk);
            }
        }
        __syncthreads();

        // (b) mixing GEMM: nxt[s][b][m] = sum_n cur[s][b][n] * mixs[m][n]
        for (int ss = 0; ss < 4; ++ss) {
            const int s = wave * 4 + ss;
            V16 af[4];
            for (int kb = 0; kb < 4; ++kb) {
                const __bf16* ap = cur + (s * 16 + lo) * PAD + kb * 32 + 8 * hi;
                af[kb].h[0] = *(const bf16x8*)(ap);
                af[kb].h[1] = *(const bf16x8*)(ap + 16);
            }
            for (int mt = 0; mt < 8; ++mt) {
                // load all 4 B fragments (one clause), then 4 chained WMMAs
                V16 bfr[4];
                for (int kb = 0; kb < 4; ++kb) {
                    const __bf16* bp = mixs + (mt * 16 + lo) * PAD + kb * 32 + 8 * hi;
                    bfr[kb].h[0] = *(const bf16x8*)(bp);
                    bfr[kb].h[1] = *(const bf16x8*)(bp + 16);
                }
                f32x8 acc;
                for (int r = 0; r < 8; ++r) acc[r] = 0.0f;
                for (int kb = 0; kb < 4; ++kb)
                    acc = WMMA_BF16(af[kb].v, bfr[kb].v, acc);
                for (int r = 0; r < 8; ++r) {
                    int b = r + 8 * hi, m = mt * 16 + lo;
                    nxt[(s * 16 + b) * PAD + m] = (__bf16)acc[r];
                }
            }
        }
        __syncthreads();

        // (c) partial batch sum for this layer's buf
        for (int p = tid; p < 4096; p += 256) {
            int m = p >> 5, s = p & 31;
            float sum = 0.0f;
            for (int b = 0; b < 16; ++b) sum += (float)nxt[(s * 16 + b) * PAD + m];
            part[((size_t)blockIdx.x * 3 + layer) * 4096 + p] = sum;
        }
        __syncthreads();

        __bf16* t = cur; cur = nxt; nxt = t;
    }

    // logits[b,n,s'] = sum_s cur[s][b][n] * readout[s', s]
    V16 rb[2];
    for (int t = 0; t < 2; ++t)
        for (int e = 0; e < 16; ++e)
            rb[t].v[e] = (__bf16)readout[(t * 16 + lo) * 32 + 16 * hi + e];

    for (int nn = 0; nn < 16; ++nn) {
        const int n = wave * 16 + nn;
        V16 a;
        for (int e = 0; e < 16; ++e) {
            int s = e + 16 * hi;
            a.v[e] = cur[(s * 16 + lo) * PAD + n];
        }
        for (int t = 0; t < 2; ++t) {
            f32x8 acc;
            for (int r = 0; r < 8; ++r) acc[r] = 0.0f;
            acc = WMMA_BF16(a.v, rb[t].v, acc);
            for (int r = 0; r < 8; ++r) {
                int b = r + 8 * hi, sp = t * 16 + lo;
                logits[((size_t)(b0 + b) * 128 + n) * 32 + sp] = acc[r];
            }
        }
    }
}

// ---------------------------------------------------------------------------
// K4a/K4b: two-stage reduction of partials (1024 WGs -> 32 -> 1)
// ---------------------------------------------------------------------------
__global__ void k_reduce1(const float* __restrict__ part, float* __restrict__ part2) {
    int j = blockIdx.x * 256 + threadIdx.x;   // 0..12287
    int g = blockIdx.y;                        // 0..31
    if (j < 12288) {
        float s = 0.0f;
        for (int w = 0; w < 32; ++w) s += part[(size_t)(g * 32 + w) * 12288 + j];
        part2[(size_t)g * 12288 + j] = s;
    }
}

__global__ void k_reduce2(const float* __restrict__ part2, float* __restrict__ bufs) {
    int j = blockIdx.x * 256 + threadIdx.x;
    if (j < 12288) {
        float s = 0.0f;
        for (int g = 0; g < 32; ++g) s += part2[(size_t)g * 12288 + j];
        bufs[j] = s * (1.0f / 16384.0f);
    }
}

// ---------------------------------------------------------------------------
// K5: recency-combine, enc/dec MLP, top-k, head -> hout[0..127]=h, hout[128]=c
// ---------------------------------------------------------------------------
__global__ void k_head(const float* __restrict__ bufs,
                       const float* __restrict__ enc_w, const float* __restrict__ enc_b,
                       const float* __restrict__ dec_w, const float* __restrict__ dec_b,
                       const float* __restrict__ hw1, const float* __restrict__ hb1,
                       const float* __restrict__ hw2, const float* __restrict__ hb2,
                       float* __restrict__ hout) {
    __shared__ float nbar[128];
    __shared__ float hsh[128];
    const int n = threadIdx.x;  // 0..127

    float nsrow[32];
    for (int s = 0; s < 32; ++s) {
        float v = 0.81f * bufs[0 * 4096 + n * 32 + s]
                + 0.90f * bufs[1 * 4096 + n * 32 + s]
                + 1.00f * bufs[2 * 4096 + n * 32 + s];
        nsrow[s] = v * (1.0f / 2.71f);
    }
    float tc[16];
    for (int c = 0; c < 16; ++c) {
        float a = enc_b[c];
        for (int s = 0; s < 32; ++s) a += nsrow[s] * enc_w[c * 32 + s];
        tc[c] = fmaxf(a, 0.0f);
    }
    float d[32];
    for (int s = 0; s < 32; ++s) {
        float a = dec_b[s];
        for (int c = 0; c < 16; ++c) a += tc[c] * dec_w[s * 16 + c];
        d[s] = a;
    }
    float ab[32];
    for (int s = 0; s < 32; ++s) ab[s] = fabsf(d[s]);
    float kth = 0.0f;
    for (int it = 0; it < 8; ++it) {
        float mx = -1.0f; int mi = 0;
        for (int s = 0; s < 32; ++s) if (ab[s] > mx) { mx = ab[s]; mi = s; }
        kth = mx; ab[mi] = -2.0f;
    }
    float msum = 0.0f;
    for (int s = 0; s < 32; ++s) msum += (fabsf(d[s]) >= kth) ? d[s] : 0.0f;
    nbar[n] = msum * (1.0f / 32.0f);
    __syncthreads();

    const int c = threadIdx.x;
    float a = hb1[c];
    for (int q = 0; q < 128; ++q) a += hw1[c * 128 + q] * nbar[q];
    float h = tanhf(a);
    hsh[c] = h;
    hout[c] = h;
    __syncthreads();
    if (c == 0) {
        float a2 = hb2[0];
        for (int q = 0; q < 128; ++q) a2 += hsh[q] * hw2[q];
        hout[128] = 1.0f / (1.0f + expf(-a2));
    }
}

// ---------------------------------------------------------------------------
// K6: broadcast p_t = h, contradiction scalar
// ---------------------------------------------------------------------------
__global__ void k_bcast(const float* __restrict__ hout, float* __restrict__ pt,
                        float* __restrict__ contra) {
    int i = blockIdx.x * 256 + threadIdx.x;
    if (i < 16384 * 128) pt[i] = hout[i & 127];
    if (i < 16384) contra[i] = hout[128];
}

// ---------------------------------------------------------------------------
extern "C" void kernel_launch(void* const* d_in, const int* in_sizes, int n_in,
                              void* d_out, int out_size, void* d_ws, size_t ws_size,
                              hipStream_t stream) {
    const float* x       = (const float*)d_in[0];
    const float* ent     = (const float*)d_in[1];
    const float* mixing  = (const float*)d_in[2];
    const float* readout = (const float*)d_in[3];
    const float* proj_w  = (const float*)d_in[4];
    const float* proj_b  = (const float*)d_in[5];
    const float* enc_w   = (const float*)d_in[6];
    const float* enc_b   = (const float*)d_in[7];
    const float* dec_w   = (const float*)d_in[8];
    const float* dec_b   = (const float*)d_in[9];
    const float* hw1     = (const float*)d_in[10];
    const float* hb1     = (const float*)d_in[11];
    const float* hw2     = (const float*)d_in[12];
    const float* hb2     = (const float*)d_in[13];

    float* out    = (float*)d_out;
    float* logits = out;                                  // 16384*128*32
    float* pt     = out + (size_t)16384 * 4096;           // 16384*128
    float* contra = pt + (size_t)16384 * 128;             // 16384

    char* ws = (char*)d_ws;
    __bf16* projbf = (__bf16*)ws;                          //   4,194,304 B
    __bf16* xbf    = (__bf16*)(ws + 4194304);              //  16,777,216 B
    __bf16* x3bf   = (__bf16*)(ws + 20971520);             // 134,217,728 B
    float*  part   = (float*)(ws + 155189248);             //  50,331,648 B
    float*  part2  = (float*)(ws + 205520896);             //   1,572,864 B
    float*  bufs   = (float*)(ws + 207093760);             //      49,152 B
    float*  hout   = (float*)(ws + 207142912);             //         516 B

    k_prep<<<32768, 256, 0, stream>>>(proj_w, projbf, x, xbf);
    k_proj<<<dim3(128, 32), 256, 128 * 520 * 2, stream>>>(xbf, projbf, proj_b, x3bf);
    k_fused<<<1024, 256, (2 * 69632 + 128 * 136) * 2, stream>>>(x3bf, ent, mixing,
                                                                readout, logits, part);
    k_reduce1<<<dim3(48, 32), 256, 0, stream>>>(part, part2);
    k_reduce2<<<48, 256, 0, stream>>>(part2, bufs);
    k_head<<<1, 128, 0, stream>>>(bufs, enc_w, enc_b, dec_w, dec_b,
                                  hw1, hb1, hw2, hb2, hout);
    k_bcast<<<8192, 256, 0, stream>>>(hout, pt, contra);
}
